// CopyModel_85426899517638
// MI455X (gfx1250) — compile-verified, matrix operand order checked by reading, not run on hardware
//
#include <hip/hip_runtime.h>
#include <hip/hip_bf16.h>

// ---------------------------------------------------------------------------
// UR-LSTM for MI455X (gfx1250): persistent recurrent kernel, bf16 WMMA,
// weights stationary in LDS (TDM-staged), gates+bias+input folded into one
// augmented GEMM, h history kept in WMMA-fragment layout end-to-end.
// ---------------------------------------------------------------------------

typedef __attribute__((ext_vector_type(16))) __bf16 v16bf;
typedef __attribute__((ext_vector_type(8)))  float  v8f;
typedef __attribute__((ext_vector_type(4)))  unsigned int u32x4;
typedef __attribute__((ext_vector_type(8)))  int    i32x8;
typedef __attribute__((ext_vector_type(4)))  int    i32x4;

union AFrag { uint4 u[2]; v16bf v; };

// Geometry
constexpr int  BB   = 128;     // batch
constexpr int  TT   = 1024;    // time
constexpr int  II   = 10;      // input dim
constexpr int  HH   = 512;     // hidden
constexpr int  NWG  = 16;      // persistent workgroups; each owns 32 hidden units
constexpr int  NT   = 8;       // n-tiles per WG (128 gate cols)
constexpr int  KT   = 17;      // k-tiles of 32: 16 for h, 1 for [x_t | 1 | pad]
constexpr int  TILE = 1024;    // bytes per 16x32 bf16 fragment tile

constexpr size_t STEPB = (size_t)16 * 8 * TILE;   // h-frag bytes per timestep (128 KB)

// Workspace layout (bytes)
constexpr size_t WS_CNT  = 0;                                   // barrier counter
constexpr size_t WS_ZERO = 4096;                                // zeroed h(-1) page
constexpr size_t WS_BUFB = WS_ZERO + STEPB;                     // packed weights
constexpr size_t SZ_BUFB = (size_t)NWG * NT * KT * TILE;        // ~2.2 MB
constexpr size_t WS_BUFX = WS_BUFB + SZ_BUFB;                   // packed x A-frags
constexpr size_t SZ_BUFX = (size_t)TT * 8 * TILE;               // 8 MB
constexpr size_t WS_WOP  = WS_BUFX + SZ_BUFX;                   // permuted W_out
constexpr size_t SZ_WOP  = (size_t)10 * 512 * 4;                // 20 KB
constexpr size_t WS_HSF  = ((WS_WOP + SZ_WOP + 4095) / 4096) * 4096; // h frags, all t (128 MB)

// K position held by (vgpr v, half h2, lane) in a 16-bit A/B fragment
// (CDNA5 ISA: lanes 0-15 hold K {0..7,16..23}, lanes 16-31 +8)
__device__ __forceinline__ int k_local_of(int v, int h2, int lane) {
  int base = (v < 4) ? (2 * v) : (16 + 2 * (v - 4));
  return base + h2 + ((lane & 16) ? 8 : 0);
}

// Inverse: byte offset inside a 1KB fragment tile for (row M, k)
__device__ __forceinline__ int frag_byte_off(int row, int k) {
  int hi   = (k >> 3) & 1;
  int k2   = k - 8 * hi;
  int h2   = k2 & 1;
  int base = k2 - h2;
  int v    = (base < 8) ? (base >> 1) : (4 + ((base - 16) >> 1));
  return (row + 16 * hi) * 32 + v * 4 + h2 * 2;
}

__device__ __forceinline__ unsigned short bf16_bits(float f) {
  union { __hip_bfloat16 h; unsigned short u; } cv;
  cv.h = __float2bfloat16(f);
  return cv.u;
}

__device__ __forceinline__ float fast_sigmoid(float x) {
#if __has_builtin(__builtin_amdgcn_rcpf)
  return __builtin_amdgcn_rcpf(1.f + __expf(-x));
#else
  return 1.f / (1.f + __expf(-x));
#endif
}

__device__ __forceinline__ float fast_tanh(float x) {
#if __has_builtin(__builtin_amdgcn_tanhf)
  return __builtin_amdgcn_tanhf(x);            // gfx1250 native v_tanh_f32
#elif __has_builtin(__builtin_amdgcn_tanh_f32)
  return __builtin_amdgcn_tanh_f32(x);
#else
  return 1.f - 2.f * fast_sigmoid(-2.f * x);   // tanh via exp
#endif
}

// ---------------------------------------------------------------------------
// Zero barrier counter + the h(-1) zero page.
// ---------------------------------------------------------------------------
__global__ void init_ws_kernel(unsigned* ws32) {
  const unsigned n = (unsigned)((WS_ZERO + STEPB) / 4);
  for (unsigned i = blockIdx.x * blockDim.x + threadIdx.x; i < n;
       i += gridDim.x * blockDim.x)
    ws32[i] = 0u;
}

// ---------------------------------------------------------------------------
// Pack augmented weights [W_hh^T ; W_ih^T ; bias ; 0] into WMMA B-fragment
// tiles, kt-major within each WG so the GEMM inner loop walks 1KB strides
// that fit DS offset immediates: tile index = (wg*KT + kt)*NT + nt.
// ---------------------------------------------------------------------------
__global__ void pack_weights_kernel(const float* __restrict__ W_hh,
                                    const float* __restrict__ W_ih,
                                    const float* __restrict__ bias,
                                    char* __restrict__ ws) {
  const int idx = blockIdx.x * blockDim.x + threadIdx.x;
  if (idx >= NWG * NT * KT * 32) return;
  const int lane = idx & 31;
  const int kt   = (idx >> 5) % KT;
  const int nt   = ((idx >> 5) / KT) % NT;
  const int wg   = (idx >> 5) / (KT * NT);
  const int q = nt >> 1, p = nt & 1, N = lane & 15;
  const int g = q * HH + wg * 32 + p * 16 + N;

  union { unsigned short s[16]; uint4 u4[2]; } pk;
#pragma unroll
  for (int v = 0; v < 8; ++v)
#pragma unroll
    for (int h2 = 0; h2 < 2; ++h2) {
      const int k = kt * 32 + k_local_of(v, h2, lane);
      float val = 0.f;
      if (k < HH)            val = W_hh[g * HH + k];
      else if (k < HH + II)  val = W_ih[g * II + (k - HH)];
      else if (k == HH + II) val = bias[g];
      pk.s[v * 2 + h2] = bf16_bits(val);
    }
  uint4* dst = (uint4*)(ws + WS_BUFB + ((size_t)((wg * KT + kt) * NT + nt)) * TILE
                        + lane * 32);
  dst[0] = pk.u4[0];
  dst[1] = pk.u4[1];
}

// ---------------------------------------------------------------------------
// Pack [x_t | 1 | 0] into A-fragment tiles for all t.
// ---------------------------------------------------------------------------
__global__ void pack_x_kernel(const float* __restrict__ x, char* __restrict__ ws) {
  const int idx = blockIdx.x * blockDim.x + threadIdx.x;
  if (idx >= TT * 8 * 32) return;
  const int lane = idx & 31;
  const int m    = (idx >> 5) & 7;
  const int t    = idx >> 8;
  const int b    = m * 16 + (lane & 15);

  union { unsigned short s[16]; uint4 u4[2]; } pk;
#pragma unroll
  for (int v = 0; v < 8; ++v)
#pragma unroll
    for (int h2 = 0; h2 < 2; ++h2) {
      const int kl = k_local_of(v, h2, lane);
      float val = 0.f;
      if (kl < II)       val = x[((size_t)b * TT + t) * II + kl];
      else if (kl == II) val = 1.0f;                       // bias column
      pk.s[v * 2 + h2] = bf16_bits(val);
    }
  uint4* dst = (uint4*)(ws + WS_BUFX + ((size_t)(t * 8 + m)) * TILE + lane * 32);
  dst[0] = pk.u4[0];
  dst[1] = pk.u4[1];
}

// ---------------------------------------------------------------------------
// Permute W_out to match fragment storage order used by the epilogue.
// ---------------------------------------------------------------------------
__global__ void pack_wout_kernel(const float* __restrict__ W_out,
                                 float* __restrict__ Wp) {
  const int idx = blockIdx.x * blockDim.x + threadIdx.x;
  if (idx >= 10 * 512) return;
  const int e = idx & 15, l = (idx >> 4) & 31, o = idx >> 9;
  const int v = e >> 1, h2 = e & 1;
  const int j = (l >> 1) * 32 + k_local_of(v, h2, (l & 1) ? 16 : 0);
  Wp[idx] = W_out[o * 512 + j];
}

// ---------------------------------------------------------------------------
// Persistent recurrent kernel: 16 WGs x 256 threads (8 waves of 32).
// ---------------------------------------------------------------------------
__global__ __launch_bounds__(256, 1)
void urlstm_persistent_kernel(const float* __restrict__ fb, char* __restrict__ ws) {
  extern __shared__ char smem[];                       // NT*KT*TILE = 136 KB
  unsigned* counter = (unsigned*)(ws + WS_CNT);
  const char* bufX  = ws + WS_BUFX;
  char* hsf         = ws + WS_HSF;

  const int wg   = blockIdx.x;
  const int tid  = threadIdx.x;
  const int m    = tid >> 5;
  const int lane = tid & 31;
  const int nlo  = lane & 15;

  // ---- Stage this WG's pre-swizzled weight slice into LDS (once) ----
#if __has_builtin(__builtin_amdgcn_tensor_load_to_lds)
  if (m == 0) {                                        // one wave issues the TDM
    const unsigned long long ga =
        (unsigned long long)(ws + WS_BUFB + (size_t)wg * NT * KT * TILE);
    const unsigned ldsa  = (unsigned)(size_t)(void*)smem;
    const unsigned nelem = (unsigned)(NT * KT * TILE / 8);   // 17408 x 8B
    // D# group0: count=1, lds_addr, global_addr[56:0], type=2 ("image")
    u32x4 g0 = { 1u, ldsa, (unsigned)(ga & 0xffffffffu),
                 (unsigned)((ga >> 32) & 0x1ffffffu) | (2u << 30) };
    // D# group1: data_size=8B, tensor_dim0=nelem, tensor_dim1=1, tile_dim0=nelem
    i32x8 g1 = { (int)(3u << 16),
                 (int)((nelem & 0xffffu) << 16),
                 (int)(((nelem >> 16) & 0xffffu) | (1u << 16)),
                 (int)(nelem << 16),
                 0,
                 (int)nelem,                                  // dim0 stride
                 0, 0 };
    i32x4 g2 = { 0, 0, 0, 0 };
    i32x4 g3 = { 0, 0, 0, 0 };
#if defined(__clang_major__) && (__clang_major__ >= 23)
    i32x8 g4 = { 0, 0, 0, 0, 0, 0, 0, 0 };
    __builtin_amdgcn_tensor_load_to_lds(g0, g1, g2, g3, g4, 0);
#else
    __builtin_amdgcn_tensor_load_to_lds(g0, g1, g2, g3, 0);
#endif
#if __has_builtin(__builtin_amdgcn_s_wait_tensorcnt)
    __builtin_amdgcn_s_wait_tensorcnt(0);
#endif
  }
#else
  {
    const uint4* src = (const uint4*)(ws + WS_BUFB + (size_t)wg * NT * KT * TILE);
    uint4* dst = (uint4*)smem;
    const int n16 = NT * KT * TILE / 16;
    for (int i = tid; i < n16; i += 256) dst[i] = src[i];
  }
#endif
  __syncthreads();

  float fbv[2];
  fbv[0] = fb[wg * 32 + nlo];
  fbv[1] = fb[wg * 32 + 16 + nlo];

  float c_state[2][8];
#pragma unroll
  for (int p = 0; p < 2; ++p)
#pragma unroll
    for (int r = 0; r < 8; ++r) c_state[p][r] = 0.f;

  const int row_base = (lane & 16) ? 8 : 0;

  for (int t = 0; t < TT; ++t) {
    v8f acc[NT];
#pragma unroll
    for (int nt = 0; nt < NT; ++nt) {
      v8f z = {0.f, 0.f, 0.f, 0.f, 0.f, 0.f, 0.f, 0.f};
      acc[nt] = z;
    }

    const char* hprev =
        (t == 0) ? (ws + WS_ZERO) : (hsf + (size_t)(t - 1) * STEPB);
    const uint4* xfrag =
        (const uint4*)(bufX + ((size_t)(t * 8 + m)) * TILE + lane * 32);

    // prefetch next step's x-fragment tile
    __builtin_prefetch((const void*)(bufX +
        ((size_t)(((t + 1) & (TT - 1)) * 8 + m)) * TILE + lane * 32), 0, 1);

    // Software-pipelined GEMM: A fragment for kt+1 loads under kt's WMMAs;
    // B fragments live in LDS at kt-major 1KB strides (DS offset immediates).
    AFrag a_cur, a_nxt;
    {
      const uint4* ap0 = (const uint4*)(hprev + (size_t)m * TILE + lane * 32);
      a_cur.u[0] = ap0[0];
      a_cur.u[1] = ap0[1];
    }
#pragma unroll
    for (int kt = 0; kt < KT; ++kt) {
      if (kt + 1 < KT) {
        const uint4* ap =
            (kt + 1 < 16)
                ? (const uint4*)(hprev + (size_t)((kt + 1) * 8 + m) * TILE + lane * 32)
                : xfrag;
        a_nxt.u[0] = ap[0];
        a_nxt.u[1] = ap[1];
      }
      const char* kbase = smem + (size_t)kt * (NT * TILE) + lane * 32;
#pragma unroll
      for (int nt = 0; nt < NT; ++nt) {
        AFrag bf;
        const uint4* bp = (const uint4*)(kbase + nt * TILE);
        bf.u[0] = bp[0];
        bf.u[1] = bp[1];
        acc[nt] = __builtin_amdgcn_wmma_f32_16x16x32_bf16(
            false, a_cur.v, false, bf.v, (short)0, acc[nt], false, false);
      }
      a_cur = a_nxt;
    }

    // ---- register-local UR-LSTM nonlinearity; single h store (frag layout) ----
    char* tile_w = hsf + (size_t)t * STEPB + (size_t)(wg * 8 + m) * TILE;
#pragma unroll
    for (int p = 0; p < 2; ++p) {
      const int jl = p * 16 + nlo;                     // col within WG slice
#pragma unroll
      for (int r = 0; r < 8; ++r) {
        const float fg = fast_sigmoid(acc[0 + p][r] + fbv[p]);
        const float rg = fast_sigmoid(acc[2 + p][r] - fbv[p]);
        const float g  = 2.f * rg * fg + (1.f - 2.f * rg) * fg * fg;
        const float cc = g * c_state[p][r] + (1.f - g) * fast_tanh(acc[4 + p][r]);
        c_state[p][r]  = cc;
        const float hv = fast_sigmoid(acc[6 + p][r]) * fast_tanh(cc);

        const int row = r + row_base;                  // b_local in [0,16)
        *(__hip_bfloat16*)(tile_w + frag_byte_off(row, jl)) = __float2bfloat16(hv);
      }
    }

    // ---- device-wide step barrier ----
    __threadfence();
    __syncthreads();
#if __has_builtin(__builtin_amdgcn_s_cluster_barrier)
    __builtin_amdgcn_s_cluster_barrier();   // cheap path if clustered; NOP otherwise
#endif
    if (tid == 0) {
      __hip_atomic_fetch_add(counter, 1u, __ATOMIC_RELEASE, __HIP_MEMORY_SCOPE_AGENT);
      const unsigned target = (unsigned)NWG * (unsigned)(t + 1);
      while (__hip_atomic_load(counter, __ATOMIC_ACQUIRE, __HIP_MEMORY_SCOPE_AGENT) < target)
        __builtin_amdgcn_s_sleep(1);
    }
    __syncthreads();
  }
}

// ---------------------------------------------------------------------------
// Epilogue: out[bt,:10] = h_t @ W_out^T + b_out, reading h directly from the
// fragment-layout history with a permuted W_out; wave32 shuffle reduction.
// ---------------------------------------------------------------------------
__global__ __launch_bounds__(256)
void out_proj_kernel(const char* __restrict__ hsf,
                     const float* __restrict__ Wp,
                     const float* __restrict__ b_out,
                     float* __restrict__ out) {
  const int gid  = blockIdx.x * 256 + threadIdx.x;
  const int wid  = gid >> 5;                 // row index b*T + t
  const int lane = gid & 31;
  if (wid >= BB * TT) return;
  const int b = wid >> 10, t = wid & (TT - 1);
  const int m = b >> 4, row = b & 15;
  const int kt = lane >> 1;                  // each lane: 16 h elems of one k-tile
  const int sl = row + 16 * (lane & 1);

  union { uint4 q[2]; __hip_bfloat16 h[16]; } hb;
  const uint4* hp = (const uint4*)(hsf + (size_t)t * STEPB +
                                   (size_t)(kt * 8 + m) * TILE + sl * 32);
  hb.q[0] = hp[0];
  hb.q[1] = hp[1];
  float hv[16];
#pragma unroll
  for (int i = 0; i < 16; ++i) hv[i] = __bfloat162float(hb.h[i]);

  float acc[10];
#pragma unroll
  for (int o = 0; o < 10; ++o) {
    const float* wr = Wp + o * 512 + lane * 16;
    float s = 0.f;
#pragma unroll
    for (int i = 0; i < 16; ++i) s += hv[i] * wr[i];
    acc[o] = s;
  }
#pragma unroll
  for (int off = 16; off > 0; off >>= 1)
#pragma unroll
    for (int o = 0; o < 10; ++o) acc[o] += __shfl_xor(acc[o], off, 32);

  if (lane == 0) {
#pragma unroll
    for (int o = 0; o < 10; ++o) out[(size_t)wid * 10 + o] = acc[o] + b_out[o];
  }
}

// ---------------------------------------------------------------------------
extern "C" void kernel_launch(void* const* d_in, const int* in_sizes, int n_in,
                              void* d_out, int out_size, void* d_ws, size_t ws_size,
                              hipStream_t stream) {
  const float* x     = (const float*)d_in[0];
  const float* W_ih  = (const float*)d_in[1];
  const float* W_hh  = (const float*)d_in[2];
  const float* bias  = (const float*)d_in[3];
  const float* fb    = (const float*)d_in[4];
  const float* W_out = (const float*)d_in[5];
  const float* b_out = (const float*)d_in[6];
  char* ws = (char*)d_ws;
  (void)in_sizes; (void)n_in; (void)out_size; (void)ws_size;

  const int lds_bytes = NT * KT * TILE;  // 136 KB dynamic LDS (gfx1250: 320 KB/WGP)
  hipFuncSetAttribute((const void*)urlstm_persistent_kernel,
                      hipFuncAttributeMaxDynamicSharedMemorySize, lds_bytes);

  init_ws_kernel<<<64, 256, 0, stream>>>((unsigned*)ws);
  pack_weights_kernel<<<(NWG * NT * KT * 32 + 255) / 256, 256, 0, stream>>>(W_hh, W_ih, bias, ws);
  pack_x_kernel<<<(TT * 8 * 32 + 255) / 256, 256, 0, stream>>>(x, ws);
  pack_wout_kernel<<<(10 * 512 + 255) / 256, 256, 0, stream>>>(W_out, (float*)(ws + WS_WOP));
  urlstm_persistent_kernel<<<NWG, 256, lds_bytes, stream>>>(fb, ws);
  out_proj_kernel<<<(BB * TT * 32 + 255) / 256, 256, 0, stream>>>(
      ws + WS_HSF, (float*)(ws + WS_WOP), b_out, (float*)d_out);
}
